// HyperGraphOptimizedAttention_81466939670772
// MI455X (gfx1250) — matrix operand order repaired and checked by reading, not compiled, not run on hardware
//
#include <hip/hip_runtime.h>
#include <hip/hip_bf16.h>

// ---------------------------------------------------------------------------
// HyperGraph routed attention for MI455X (gfx1250, wave32, WMMA)
// All heavy math runs through v_wmma_f32_16x16x32_bf16 (f32 accumulate).
// ---------------------------------------------------------------------------

typedef unsigned short u16;
typedef unsigned int   u32;
typedef __attribute__((ext_vector_type(16))) __bf16        v16bf;
typedef __attribute__((ext_vector_type(8)))  float         v8f;
typedef __attribute__((ext_vector_type(8)))  unsigned int  v8u;

union Frag { v8u u; v16bf b; };

#define HGA_N    4096
#define HGA_E    2048
#define HGA_H    16
#define HGA_D    128
#define HGA_K    4
#define HGA_CAP  1280
#define HGA_HD   2048
#define HGA_SCALE 0.08838834764831845f   // 128^-0.5
#define HGA_NEGINF (-1.0e30f)

__device__ __forceinline__ u16 f2bf(float f) {
  u32 u = __float_as_uint(f);
  u32 r = u + 0x7FFFu + ((u >> 16) & 1u);   // round-to-nearest-even
  return (u16)(r >> 16);
}
__device__ __forceinline__ float bf2f(u16 h) {
  return __uint_as_float(((u32)h) << 16);
}

// ---------------------------------------------------------------------------
// fp32 -> bf16 cast (grid-stride)
// ---------------------------------------------------------------------------
__global__ void hga_cast(const float* __restrict__ s, u16* __restrict__ d, size_t n) {
  size_t i  = (size_t)blockIdx.x * blockDim.x + threadIdx.x;
  size_t st = (size_t)gridDim.x * blockDim.x;
  for (; i < n; i += st) d[i] = f2bf(s[i]);
}

// ---------------------------------------------------------------------------
// Tiled bf16 WMMA GEMM: C[M,N] = act(A[M,K] * B[K,N] + bias)
// Block: 256 threads (8 waves). Block tile 128x128, BK=64 (two WMMA K-steps
// per stage -> 16 WMMAs between barriers). Wave (wm,wn) owns a 32x64 tile.
// Template params kill all epilogue branching per instantiation.
// ---------------------------------------------------------------------------
template <int ACT, bool HAS_BIAS, bool OUT_F32, bool OUT_BF16>
__global__ __launch_bounds__(256) void hga_gemm(
    const u16* __restrict__ A, const u16* __restrict__ B,
    const float* __restrict__ bias, float* __restrict__ Cf, u16* __restrict__ Cb,
    int M, int N, int Kd) {
  __shared__ u32 AsU[128 * 32];            // A tile 128x64 bf16, row-major
  __shared__ u32 BsU[128 * 32];            // B tile transposed: [n 0..127][k 0..63]

  const int tid  = threadIdx.x;
  const int lane = tid & 31;
  const int wave = tid >> 5;
  const int wm   = wave & 3;               // 32-row sub-block
  const int wn   = wave >> 2;              // 64-col sub-block
  const int m0   = blockIdx.y * 128;
  const int n0   = blockIdx.x * 128;
  const int l15  = lane & 15;
  const int hi   = (lane < 16) ? 0 : 1;

  v8f acc[2][4];
  const v8f vz = {0.f,0.f,0.f,0.f,0.f,0.f,0.f,0.f};
  #pragma unroll
  for (int i = 0; i < 2; ++i)
    #pragma unroll
    for (int j = 0; j < 4; ++j) acc[i][j] = vz;

  const int ar = tid >> 1;                 // A staging: row 0..127
  const int ah = (tid & 1) * 16;           // dword half of 32-dword row
  const int bk = tid & 63;                 // B staging: k 0..63
  const int bn = (tid >> 6) * 32;          // B staging: 32-col group
  const u32* Ag = (const u32*)A;
  const u32* Bg = (const u32*)B;
  u16* Bs16 = (u16*)BsU;

  for (int kc = 0; kc < Kd; kc += 64) {
    __syncthreads();
    // stage A (coalesced dword loads, vectorized by compiler to b128)
    size_t abase = (((size_t)(m0 + ar) * Kd + kc) >> 1) + ah;
    #pragma unroll
    for (int j = 0; j < 16; ++j) AsU[ar * 32 + ah + j] = Ag[abase + j];
    // stage B transposed
    size_t bbase = ((size_t)(kc + bk) * N + n0 + bn) >> 1;
    #pragma unroll
    for (int j = 0; j < 16; ++j) {
      u32 w = Bg[bbase + j];
      int c0 = bn + 2 * j;
      Bs16[c0 * 64 + bk]       = (u16)w;
      Bs16[(c0 + 1) * 64 + bk] = (u16)(w >> 16);
    }
    // prefetch next tiles into cache (global_prefetch_b8)
    if (kc + 64 < Kd) {
      __builtin_prefetch(Ag + abase + 32, 0, 3);
      __builtin_prefetch(Bg + bbase + (size_t)32 * N, 0, 3);
    }
    __syncthreads();

    #pragma unroll
    for (int ks = 0; ks < 2; ++ks) {
      // A fragments (16x32 bf16, ISA pair layout)
      Frag af[2];
      #pragma unroll
      for (int ms = 0; ms < 2; ++ms) {
        int row = wm * 32 + ms * 16 + l15;
        #pragma unroll
        for (int j = 0; j < 8; ++j) {
          int kd = ((j < 4) ? j : j + 4) + hi * 4;
          af[ms].u[j] = AsU[row * 32 + ks * 16 + kd];
        }
      }
      // B fragments (32x16 bf16, ISA pair layout matching A)
      Frag bf[4];
      #pragma unroll
      for (int ns = 0; ns < 4; ++ns) {
        int n = wn * 64 + ns * 16 + l15;
        #pragma unroll
        for (int j = 0; j < 8; ++j) bf[ns].u[j] = BsU[n * 32 + ks * 16 + ((j < 4) ? j : j + 4) + hi * 4];
      }
      #pragma unroll
      for (int ms = 0; ms < 2; ++ms)
        #pragma unroll
        for (int ns = 0; ns < 4; ++ns)
          acc[ms][ns] = __builtin_amdgcn_wmma_f32_16x16x32_bf16(
              false, af[ms].b, false, bf[ns].b, (short)0, acc[ms][ns], false, false);
    }
  }

  // branchless epilogue (templated)
  #pragma unroll
  for (int ms = 0; ms < 2; ++ms) {
    #pragma unroll
    for (int ns = 0; ns < 4; ++ns) {
      int gn = n0 + wn * 64 + ns * 16 + l15;
      float bv = 0.f;
      if (HAS_BIAS) bv = bias[gn];
      #pragma unroll
      for (int j = 0; j < 8; ++j) {
        int gm = m0 + wm * 32 + ms * 16 + j + hi * 8;
        float v = acc[ms][ns][j] + bv;
        if (ACT == 1) v = 0.5f * v * (1.0f + erff(v * 0.70710678118654752f));
        size_t o = (size_t)gm * N + gn;
        if (OUT_F32)  Cf[o] = v;
        if (OUT_BF16) Cb[o] = f2bf(v);
      }
    }
  }
}

// ---------------------------------------------------------------------------
// Router scores: scores[n,k] = h[n,:] @ Wr2 + br2 ; per-block partial of
// sum_n max_k(scores) for the aux loss (deterministic: ordered partials).
// ---------------------------------------------------------------------------
__global__ __launch_bounds__(256) void hga_scores(
    const u16* __restrict__ hb, const float* __restrict__ Wr2,
    const float* __restrict__ br2, float* __restrict__ scores,
    float* __restrict__ auxp) {
  __shared__ float w2[1024 * 4];
  __shared__ float red[256];
  const int tid = threadIdx.x;
  for (int i = tid; i < 4096; i += 256) w2[i] = Wr2[i];
  __syncthreads();

  const int tok = blockIdx.x * 256 + tid;
  float a0 = br2[0], a1 = br2[1], a2 = br2[2], a3 = br2[3];
  const u32* hr = (const u32*)(hb + (size_t)tok * 1024);
  for (int j = 0; j < 512; ++j) {
    u32 w = hr[j];
    float h0 = bf2f((u16)w), h1 = bf2f((u16)(w >> 16));
    int j0 = 2 * j, j1 = 2 * j + 1;
    a0 += h0 * w2[j0 * 4 + 0] + h1 * w2[j1 * 4 + 0];
    a1 += h0 * w2[j0 * 4 + 1] + h1 * w2[j1 * 4 + 1];
    a2 += h0 * w2[j0 * 4 + 2] + h1 * w2[j1 * 4 + 2];
    a3 += h0 * w2[j0 * 4 + 3] + h1 * w2[j1 * 4 + 3];
  }
  scores[tok * 4 + 0] = a0; scores[tok * 4 + 1] = a1;
  scores[tok * 4 + 2] = a2; scores[tok * 4 + 3] = a3;
  red[tid] = fmaxf(fmaxf(a0, a1), fmaxf(a2, a3));
  __syncthreads();
  for (int st = 128; st > 0; st >>= 1) {
    if (tid < st) red[tid] += red[tid + st];
    __syncthreads();
  }
  if (tid == 0) auxp[blockIdx.x] = red[0];
}

// ---------------------------------------------------------------------------
// Per-timeline top-CAP via full bitonic sort of 4096 (score,idx) pairs in LDS,
// then softmax over the kept 1280. One 1024-thread workgroup per timeline.
// ---------------------------------------------------------------------------
__global__ __launch_bounds__(1024) void hga_topk(
    const float* __restrict__ scores, int* __restrict__ tidx,
    float* __restrict__ selw) {
  __shared__ float s[4096];
  __shared__ int   si[4096];
  __shared__ float red[1024];
  const int t = blockIdx.x;
  const int tid = threadIdx.x;
  for (int i = tid; i < 4096; i += 1024) { s[i] = scores[i * 4 + t]; si[i] = i; }
  __syncthreads();
  for (int ksz = 2; ksz <= 4096; ksz <<= 1) {
    for (int j = ksz >> 1; j > 0; j >>= 1) {
      for (int i = tid; i < 4096; i += 1024) {
        int ixj = i ^ j;
        if (ixj > i) {
          float sa = s[i], sb = s[ixj];
          int   ia = si[i], ib = si[ixj];
          bool aGreater = (sa > sb) || (sa == sb && ia < ib);
          bool wantDesc = (i & ksz) == 0;
          if (wantDesc ? (!aGreater) : aGreater) {
            s[i] = sb; s[ixj] = sa; si[i] = ib; si[ixj] = ia;
          }
        }
      }
      __syncthreads();
    }
  }
  float mx = s[0];
  float part = 0.f;
  for (int i = tid; i < HGA_CAP; i += 1024) part += __expf(s[i] - mx);
  red[tid] = part;
  __syncthreads();
  for (int st = 512; st > 0; st >>= 1) {
    if (tid < st) red[tid] += red[tid + st];
    __syncthreads();
  }
  float inv = 1.0f / red[0];
  for (int i = tid; i < HGA_CAP; i += 1024) {
    tidx[t * HGA_CAP + i] = si[i];
    selw[t * HGA_CAP + i] = __expf(s[i] - mx) * inv;
  }
}

// ---------------------------------------------------------------------------
// Gather selected tokens, apply RoPE at timeline position p, emit bf16 q/k/v
// laid out [t][h][p][d]. One block per (t,p), 256 threads over H*D.
// ---------------------------------------------------------------------------
__global__ __launch_bounds__(256) void hga_gather_rope(
    const float* __restrict__ qf, const float* __restrict__ kf,
    const float* __restrict__ vf, const int* __restrict__ tidx,
    u16* __restrict__ qt, u16* __restrict__ kt, u16* __restrict__ vt) {
  const int b = blockIdx.x;
  const int t = b / HGA_CAP;
  const int p = b % HGA_CAP;
  const int tok = tidx[t * HGA_CAP + p];
  const size_t src = (size_t)tok * HGA_HD;
  for (int e = threadIdx.x; e < HGA_HD; e += 256) {
    int h = e >> 7, dd = e & 127;
    int jj = dd & 63;
    float ang = (float)p * __powf(10000.0f, -(float)jj * (1.0f / 64.0f));
    float sn, cs;
    __sincosf(ang, &sn, &cs);
    int   e2  = (dd < 64) ? e + 64 : e - 64;
    float sgn = (dd < 64) ? -1.0f : 1.0f;
    float rq = qf[src + e] * cs + sgn * qf[src + e2] * sn;
    float rk = kf[src + e] * cs + sgn * kf[src + e2] * sn;
    size_t o = ((size_t)(t * HGA_H + h) * HGA_CAP + p) * HGA_D + dd;
    qt[o] = f2bf(rq);
    kt[o] = f2bf(rk);
    vt[o] = f2bf(vf[src + e]);
  }
}

// ---------------------------------------------------------------------------
// Flash attention over one timeline/head. One wave (32 threads) per 16-row
// q-tile; online softmax over 32-key chunks; QK^T and P*V via bf16 WMMA.
// All 8 K fragments are preloaded into distinct registers so the 16 b128
// loads issue as one clause and the 8 S-tile WMMAs drain with staged waits.
// ---------------------------------------------------------------------------
__global__ __launch_bounds__(32) void hga_flash(
    const u16* __restrict__ qt, const u16* __restrict__ kt,
    const u16* __restrict__ vt, const float* __restrict__ selw,
    float* __restrict__ ot) {
  const int qb   = blockIdx.x;                  // 0..79
  const int h    = blockIdx.y;                  // 0..15
  const int t    = blockIdx.z;                  // 0..3
  const int lane = threadIdx.x;
  const int l15  = lane & 15;
  const int hi   = (lane < 16) ? 0 : 1;
  const int q0   = qb * 16;
  const size_t base = (size_t)(t * HGA_H + h) * HGA_CAP * HGA_D;  // u16 elems
  const u32* Q  = ((const u32*)qt) + (base >> 1);
  const u32* Kp = ((const u32*)kt) + (base >> 1);
  const u32* Vp = ((const u32*)vt) + (base >> 1);

  __shared__ float ps[16 * 32];                 // P tile (f32, C layout dump)
  __shared__ u16   vts[128 * 32];               // V^T chunk: [d][kd]

  // Q fragments: 4 chunks of K-dim (D=128); two aligned b128 loads per frag
  Frag aq[4];
  {
    int row = q0 + l15;
    #pragma unroll
    for (int c = 0; c < 4; ++c) {
      const uint4* qp = (const uint4*)&Q[row * 64 + c * 16 + hi * 4];
      uint4 w0 = qp[0];        // dwords kd 0..3 (+hi*4)
      uint4 w1 = qp[2];        // dwords kd 8..11 (+hi*4)
      aq[c].u[0] = w0.x; aq[c].u[1] = w0.y; aq[c].u[2] = w0.z; aq[c].u[3] = w0.w;
      aq[c].u[4] = w1.x; aq[c].u[5] = w1.y; aq[c].u[6] = w1.z; aq[c].u[7] = w1.w;
    }
  }

  const v8f vz = {0.f,0.f,0.f,0.f,0.f,0.f,0.f,0.f};
  v8f o[8];
  #pragma unroll
  for (int dt = 0; dt < 8; ++dt) o[dt] = vz;
  float mrow[8], lrow[8];
  #pragma unroll
  for (int j = 0; j < 8; ++j) { mrow[j] = -3.0e38f; lrow[j] = 0.f; }

  const int nch = (q0 + 15) / 32 + 1;           // causal 32-key chunks
  for (int kc = 0; kc < nch; ++kc) {
    const int kb = kc * 32;

    // --- preload ALL K fragments for this chunk (distinct registers) ------
    Frag bkf[2][4];
    {
      const uint4* kb0 = (const uint4*)Kp + (size_t)(kb + l15) * 16 + hi;
      const uint4* kb1 = (const uint4*)Kp + (size_t)(kb + 16 + l15) * 16 + hi;
      #pragma unroll
      for (int c = 0; c < 4; ++c) {
        uint4 a0 = kb0[c * 4 + 0], a1 = kb0[c * 4 + 2];
        uint4 c0 = kb1[c * 4 + 0], c1 = kb1[c * 4 + 2];
        bkf[0][c].u[0]=a0.x; bkf[0][c].u[1]=a0.y; bkf[0][c].u[2]=a0.z; bkf[0][c].u[3]=a0.w;
        bkf[0][c].u[4]=a1.x; bkf[0][c].u[5]=a1.y; bkf[0][c].u[6]=a1.z; bkf[0][c].u[7]=a1.w;
        bkf[1][c].u[0]=c0.x; bkf[1][c].u[1]=c0.y; bkf[1][c].u[2]=c0.z; bkf[1][c].u[3]=c0.w;
        bkf[1][c].u[4]=c1.x; bkf[1][c].u[5]=c1.y; bkf[1][c].u[6]=c1.z; bkf[1][c].u[7]=c1.w;
      }
    }

    __syncthreads();
    // stage V^T for this chunk (b128 loads): lane <-> key row
    {
      const uint4* vr = (const uint4*)(Vp + (size_t)(kb + lane) * 64);
      #pragma unroll 4
      for (int q4 = 0; q4 < 16; ++q4) {
        uint4 w = vr[q4];
        int d0 = q4 * 8;
        vts[(d0 + 0) * 32 + lane] = (u16)w.x;
        vts[(d0 + 1) * 32 + lane] = (u16)(w.x >> 16);
        vts[(d0 + 2) * 32 + lane] = (u16)w.y;
        vts[(d0 + 3) * 32 + lane] = (u16)(w.y >> 16);
        vts[(d0 + 4) * 32 + lane] = (u16)w.z;
        vts[(d0 + 5) * 32 + lane] = (u16)(w.z >> 16);
        vts[(d0 + 6) * 32 + lane] = (u16)w.w;
        vts[(d0 + 7) * 32 + lane] = (u16)(w.w >> 16);
      }
    }
    // prefetch next chunk's K and V rows (global_prefetch_b8)
    if (kc + 1 < nch) {
      __builtin_prefetch(Kp + (size_t)(kb + 32 + lane) * 64, 0, 3);
      __builtin_prefetch(Vp + (size_t)(kb + 32 + lane) * 64, 0, 3);
    }
    // S = Q K^T (16x32): two 16x16 WMMA columns, accumulate over D
    v8f s0 = vz, s1 = vz;
    #pragma unroll
    for (int c = 0; c < 4; ++c) {
      s0 = __builtin_amdgcn_wmma_f32_16x16x32_bf16(false, aq[c].b, false, bkf[0][c].b,
                                                   (short)0, s0, false, false);
      s1 = __builtin_amdgcn_wmma_f32_16x16x32_bf16(false, aq[c].b, false, bkf[1][c].b,
                                                   (short)0, s1, false, false);
    }
    // causal mask + online softmax (row reductions within 16-lane halves)
    #pragma unroll
    for (int j = 0; j < 8; ++j) {
      int rg  = q0 + j + hi * 8;
      int k0i = kb + l15, k1i = kb + 16 + l15;
      float v0 = (k0i <= rg) ? s0[j] * HGA_SCALE : HGA_NEGINF;
      float v1 = (k1i <= rg) ? s1[j] * HGA_SCALE : HGA_NEGINF;
      float mx = fmaxf(v0, v1);
      #pragma unroll
      for (int m = 1; m < 16; m <<= 1) mx = fmaxf(mx, __shfl_xor(mx, m, 16));
      float mn    = fmaxf(mrow[j], mx);
      float alpha = __expf(mrow[j] - mn);
      float p0 = __expf(v0 - mn), p1 = __expf(v1 - mn);
      float rs = p0 + p1;
      #pragma unroll
      for (int m = 1; m < 16; m <<= 1) rs += __shfl_xor(rs, m, 16);
      lrow[j] = lrow[j] * alpha + rs;
      mrow[j] = mn;
      #pragma unroll
      for (int dt = 0; dt < 8; ++dt) o[dt][j] *= alpha;
      int row = j + hi * 8;
      ps[row * 32 + l15]      = p0;
      ps[row * 32 + 16 + l15] = p1;
    }
    __syncthreads();
    // rotate P (C layout) -> A layout bf16 fragment
    Frag ap;
    #pragma unroll
    for (int j = 0; j < 8; ++j) {
      int kk = ((j < 4) ? 2 * j : 16 + 2 * (j - 4)) + hi * 8;
      u16 lo = f2bf(ps[l15 * 32 + kk]);
      u16 hh = f2bf(ps[l15 * 32 + kk + 1]);
      ap.u[j] = (u32)lo | ((u32)hh << 16);
    }
    // O += P * V
    const u32* vtsU = (const u32*)vts;
    #pragma unroll
    for (int dt = 0; dt < 8; ++dt) {
      Frag bv;
      int d = dt * 16 + l15;
      #pragma unroll
      for (int j = 0; j < 8; ++j) bv.u[j] = vtsU[d * 16 + ((j < 4) ? j : j + 4) + hi * 4];
      o[dt] = __builtin_amdgcn_wmma_f32_16x16x32_bf16(false, ap.b, false, bv.b,
                                                      (short)0, o[dt], false, false);
    }
  }
  // epilogue: normalize, apply timeline weight, store f32
  float invl[8];
  #pragma unroll
  for (int j = 0; j < 8; ++j) {
    int row = q0 + j + hi * 8;
    invl[j] = selw[t * HGA_CAP + row] / lrow[j];
  }
  #pragma unroll
  for (int dt = 0; dt < 8; ++dt)
    #pragma unroll
    for (int j = 0; j < 8; ++j) {
      int row = q0 + j + hi * 8;
      ot[base + (size_t)row * HGA_D + dt * 16 + l15] = o[dt][j] * invl[j];
    }
}

// ---------------------------------------------------------------------------
// Per-timeline scatter-add (indices unique inside a timeline -> no atomics,
// deterministic across launches). One block per position p.
// ---------------------------------------------------------------------------
__global__ __launch_bounds__(256) void hga_scatter(
    const float* __restrict__ ot, const int* __restrict__ tidx,
    float* __restrict__ ctx, int t) {
  const int p = blockIdx.x;
  const int tok = tidx[t * HGA_CAP + p];
  for (int e = threadIdx.x; e < HGA_HD; e += 256) {
    int h = e >> 7, dd = e & 127;
    ctx[(size_t)tok * HGA_HD + e] +=
        ot[((size_t)(t * HGA_H + h) * HGA_CAP + p) * HGA_D + dd];
  }
}

// ---------------------------------------------------------------------------
// Aux-loss finalize: sum ordered partials, write -mean to d_out[N*E].
// ---------------------------------------------------------------------------
__global__ void hga_final_aux(const float* __restrict__ auxp, float* __restrict__ dst) {
  if (threadIdx.x == 0 && blockIdx.x == 0) {
    float s = 0.f;
    for (int i = 0; i < 16; ++i) s += auxp[i];
    dst[0] = -s * (1.0f / (float)HGA_N);
  }
}

// ---------------------------------------------------------------------------
// Host launcher
// ---------------------------------------------------------------------------
extern "C" void kernel_launch(void* const* d_in, const int* in_sizes, int n_in,
                              void* d_out, int out_size, void* d_ws, size_t ws_size,
                              hipStream_t stream) {
  (void)in_sizes; (void)n_in; (void)out_size; (void)ws_size;
  const float* x   = (const float*)d_in[0];
  const float* Wq  = (const float*)d_in[1];
  const float* Wk  = (const float*)d_in[2];
  const float* Wv  = (const float*)d_in[3];
  const float* Wo  = (const float*)d_in[4];
  const float* Wr1 = (const float*)d_in[5];
  const float* br1 = (const float*)d_in[6];
  const float* Wr2 = (const float*)d_in[7];
  const float* br2 = (const float*)d_in[8];
  float* out = (float*)d_out;

  const size_t NE  = (size_t)HGA_N * HGA_E;                    // 8,388,608
  const size_t TL  = (size_t)HGA_K * HGA_H * HGA_CAP * HGA_D;  // timeline elems

  char* ws = (char*)d_ws;
  size_t off = 0;
  auto alloc = [&](size_t bytes) -> char* {
    char* p = ws + off;
    off += (bytes + 255) & ~(size_t)255;
    return p;
  };
  u16*   xb    = (u16*)  alloc(NE * 2);
  u16*   wqb   = (u16*)  alloc(NE * 2);
  u16*   wkb   = (u16*)  alloc(NE * 2);
  u16*   wvb   = (u16*)  alloc(NE * 2);
  u16*   wob   = (u16*)  alloc(NE * 2);
  u16*   wr1b  = (u16*)  alloc((size_t)HGA_E * (HGA_E / 2) * 2);
  u16*   hb    = (u16*)  alloc((size_t)HGA_N * (HGA_E / 2) * 2);
  float* scr   = (float*)alloc((size_t)HGA_N * HGA_K * 4);
  float* auxp  = (float*)alloc(16 * 4);
  int*   tidx  = (int*)  alloc((size_t)HGA_K * HGA_CAP * 4);
  float* selw  = (float*)alloc((size_t)HGA_K * HGA_CAP * 4);
  float* qf    = (float*)alloc(NE * 4);
  float* kf    = (float*)alloc(NE * 4);
  float* vf    = (float*)alloc(NE * 4);
  u16*   qtb   = (u16*)  alloc(TL * 2);
  u16*   ktb   = (u16*)  alloc(TL * 2);
  u16*   vtb   = (u16*)  alloc(TL * 2);
  float* otb   = (float*)alloc(TL * 4);
  float* ctx   = (float*)alloc(NE * 4);
  u16*   ctxb  = (u16*)  alloc(NE * 2);

  // bf16 casts of activations and weights
  hga_cast<<<2048, 256, 0, stream>>>(x,   xb,   NE);
  hga_cast<<<2048, 256, 0, stream>>>(Wq,  wqb,  NE);
  hga_cast<<<2048, 256, 0, stream>>>(Wk,  wkb,  NE);
  hga_cast<<<2048, 256, 0, stream>>>(Wv,  wvb,  NE);
  hga_cast<<<2048, 256, 0, stream>>>(Wo,  wob,  NE);
  hga_cast<<<2048, 256, 0, stream>>>(Wr1, wr1b, (size_t)HGA_E * (HGA_E / 2));

  // router hidden: h = gelu(x @ Wr1 + br1)  [4096 x 1024], bf16 out
  {
    dim3 g((HGA_E / 2) / 128, HGA_N / 128);
    hga_gemm<1, true, false, true><<<g, 256, 0, stream>>>(
        xb, wr1b, br1, nullptr, hb, HGA_N, HGA_E / 2, HGA_E);
  }
  // scores + aux partials, top-k + softmax weights
  hga_scores<<<HGA_N / 256, 256, 0, stream>>>(hb, Wr2, br2, scr, auxp);
  hga_topk<<<HGA_K, 1024, 0, stream>>>(scr, tidx, selw);

  // q/k/v projections [4096 x 2048], f32 out
  {
    dim3 g(HGA_HD / 128, HGA_N / 128);
    hga_gemm<0, false, true, false><<<g, 256, 0, stream>>>(
        xb, wqb, nullptr, qf, nullptr, HGA_N, HGA_HD, HGA_E);
    hga_gemm<0, false, true, false><<<g, 256, 0, stream>>>(
        xb, wkb, nullptr, kf, nullptr, HGA_N, HGA_HD, HGA_E);
    hga_gemm<0, false, true, false><<<g, 256, 0, stream>>>(
        xb, wvb, nullptr, vf, nullptr, HGA_N, HGA_HD, HGA_E);
  }

  // gather + RoPE into bf16 timelines
  hga_gather_rope<<<HGA_K * HGA_CAP, 256, 0, stream>>>(qf, kf, vf, tidx,
                                                       qtb, ktb, vtb);
  // flash attention per (q-tile, head, timeline)
  {
    dim3 g(HGA_CAP / 16, HGA_H, HGA_K);
    hga_flash<<<g, 32, 0, stream>>>(qtb, ktb, vtb, selw, otb);
  }

  // scatter back to token axis (4 deterministic per-timeline passes)
  hipMemsetAsync(ctx, 0, NE * 4, stream);
  for (int t = 0; t < HGA_K; ++t)
    hga_scatter<<<HGA_CAP, 256, 0, stream>>>(otb, tidx, ctx, t);

  // out = ctx @ Wo  [4096 x 2048], f32 directly into d_out
  hga_cast<<<2048, 256, 0, stream>>>(ctx, ctxb, NE);
  {
    dim3 g(HGA_E / 128, HGA_N / 128);
    hga_gemm<0, false, true, false><<<g, 256, 0, stream>>>(
        ctxb, wob, nullptr, out, nullptr, HGA_N, HGA_E, HGA_HD);
  }
  // aux loss scalar
  hga_final_aux<<<1, 32, 0, stream>>>(auxp, out + NE);
}
